// attn_block_8005819039830
// MI455X (gfx1250) — matrix operand order repaired and checked by reading, not compiled, not run on hardware
//
#include <hip/hip_runtime.h>

#define B_  16
#define N_  1024
#define C_  512
#define G_  32
#define CG_ 16
#define EPS_ 1e-3f

typedef __bf16 bf16_t;
typedef __attribute__((ext_vector_type(16))) __bf16 v16bf;
typedef __attribute__((ext_vector_type(8)))  __bf16 v8bf;
typedef __attribute__((ext_vector_type(8)))  float  v8f;

union V16 { v16bf v; v8bf h[2]; };

static __device__ __forceinline__ v8f wmma_bf16(v16bf a, v16bf b, v8f c) {
  // D = A(16x32 bf16) x B(32x16 bf16) + C(16x16 f32)
  return __builtin_amdgcn_wmma_f32_16x16x32_bf16(false, a, false, b, (short)0, c, false, false);
}

// Async global->LDS copy (16B per lane), tracked by ASYNCcnt (cdna5 §10/§15.18).
// LDS aperture: addr[31:0] == LDS byte offset, so truncate the generic pointer.
static __device__ __forceinline__ void async_load_b128(void* lds_dst, const void* gsrc) {
  unsigned lds = (unsigned)(uintptr_t)lds_dst;
  asm volatile("global_load_async_to_lds_b128 %0, %1, off"
               :: "v"(lds), "v"(gsrc) : "memory");
}
static __device__ __forceinline__ void wait_async0() {
  asm volatile("s_wait_asynccnt 0x0" ::: "memory");
}

// ---------------------------------------------------------------------------
// 1) Convert 4 weight matrices f32[in][out] -> bf16 transposed wT[out][in]
// ---------------------------------------------------------------------------
__global__ void cvt_weights_kernel(const float* __restrict__ w0, const float* __restrict__ w1,
                                   const float* __restrict__ w2, const float* __restrict__ w3,
                                   bf16_t* __restrict__ o0, bf16_t* __restrict__ o1,
                                   bf16_t* __restrict__ o2, bf16_t* __restrict__ o3) {
  int i = blockIdx.x * blockDim.x + threadIdx.x;   // 0 .. 262143
  const float* w = (blockIdx.y == 0) ? w0 : (blockIdx.y == 1) ? w1 :
                   (blockIdx.y == 2) ? w2 : w3;
  bf16_t* o = (blockIdx.y == 0) ? o0 : (blockIdx.y == 1) ? o1 :
              (blockIdx.y == 2) ? o2 : o3;
  int oc = i >> 9;      // output channel
  int ic = i & 511;     // input channel
  o[(size_t)oc * C_ + ic] = (bf16_t)w[(size_t)ic * C_ + oc];
}

// ---------------------------------------------------------------------------
// 2) GroupNorm: block per (batch, group); output bf16 normalized activations
// ---------------------------------------------------------------------------
__global__ void __launch_bounds__(256)
groupnorm_kernel(const float* __restrict__ x, const float* __restrict__ gamma,
                 const float* __restrict__ beta, bf16_t* __restrict__ xn) {
  __shared__ float s_sum[256], s_sq[256];
  int tid = threadIdx.x;
  int b = blockIdx.x >> 5;
  int g = blockIdx.x & 31;
  const float* xb = x + (size_t)b * N_ * C_ + g * CG_;
  float sum = 0.f, sq = 0.f;
  for (int i = tid; i < N_ * CG_; i += 256) {
    int n = i >> 4, c = i & 15;
    float v = xb[(size_t)n * C_ + c];
    sum += v; sq += v * v;
  }
  s_sum[tid] = sum; s_sq[tid] = sq;
  __syncthreads();
  for (int st = 128; st > 0; st >>= 1) {
    if (tid < st) { s_sum[tid] += s_sum[tid + st]; s_sq[tid] += s_sq[tid + st]; }
    __syncthreads();
  }
  const float inv_n = 1.f / (float)(N_ * CG_);
  float mean = s_sum[0] * inv_n;
  float var  = s_sq[0] * inv_n - mean * mean;
  float rstd = rsqrtf(var + EPS_);
  bf16_t* xnb = xn + (size_t)b * N_ * C_ + g * CG_;
  for (int i = tid; i < N_ * CG_; i += 256) {
    int n = i >> 4, c = i & 15;
    float v = xb[(size_t)n * C_ + c];
    float y = (v - mean) * rstd * gamma[g * CG_ + c] + beta[g * CG_ + c];
    xnb[(size_t)n * C_ + c] = (bf16_t)y;
  }
}

// ---------------------------------------------------------------------------
// 3) GEMM: Y[M,512] = xn[M,512] @ W + b  (optionally scaled / transposed out)
//    8 waves/block; wave = 16x64 tile; K-step 32 -> 64 WMMA per wave.
//    Weight tile (64 cols x 512 K = 64 KB) staged in LDS once via async copy;
//    shared by all 8 waves (8x traffic reduction, ds-latency B fragments).
// ---------------------------------------------------------------------------
__global__ void __launch_bounds__(256)
qkv_gemm_kernel(const bf16_t* __restrict__ xn, const bf16_t* __restrict__ wT,
                const float* __restrict__ bias, bf16_t* __restrict__ out,
                float scale, int transposeOut) {
  __shared__ __align__(16) bf16_t w_lds[64][C_];   // 64 KB
  int tid  = threadIdx.x;
  int lane = tid & 31;
  int wave = tid >> 5;
  int mt = blockIdx.x >> 3;            // 0..127 (row blocks of 128)
  int nt = blockIdx.x & 7;             // 0..7   (col blocks of 64)
  int row0 = mt * 128 + wave * 16;
  int col0 = nt * 64;
  int lm   = lane & 15;
  int half = lane >> 4;
  int koffA = half * 16;
  int koffB = half * 16;

  // async-stage the 64x512 bf16 weight tile: 16 x b128 per thread
  const bf16_t* wsrc = wT + (size_t)col0 * C_;
#pragma unroll
  for (int it = 0; it < 16; ++it) {
    int i = it * 256 + tid;            // i-th 8-element chunk
    int row = i >> 6;                  // (i*8)/512
    int col = (i * 8) & 511;
    async_load_b128(&w_lds[row][col], wsrc + (size_t)row * C_ + col);
  }
  wait_async0();
  __syncthreads();

  const bf16_t* arow = xn + (size_t)(row0 + lm) * C_ + koffA;
  v8f acc[4] = {};
#pragma unroll
  for (int kk = 0; kk < C_; kk += 32) {
    V16 a;
    a.h[0] = *(const v8bf*)(arow + kk);
    a.h[1] = *(const v8bf*)(arow + kk + 8);
#pragma unroll
    for (int t = 0; t < 4; ++t) {
      v16bf bm = *(const v16bf*)&w_lds[t * 16 + lm][kk + koffB];
      acc[t] = wmma_bf16(a.v, bm, acc[t]);
    }
  }

#pragma unroll
  for (int t = 0; t < 4; ++t) {
    int col = col0 + t * 16 + lm;
    float bv = bias[col];
#pragma unroll
    for (int r = 0; r < 8; ++r) {
      int row = row0 + r + half * 8;
      float y = (acc[t][r] + bv) * scale;
      if (transposeOut) {          // V: store [b][c][n] for contiguous P@V B-operand
        int bb = row >> 10, n = row & 1023;
        out[((size_t)bb * C_ + col) * N_ + n] = (bf16_t)y;
      } else {
        out[(size_t)row * C_ + col] = (bf16_t)y;
      }
    }
  }
}

// ---------------------------------------------------------------------------
// 4) Fused flash-attention + output projection + residual
//    1 block = (batch b, 16-query tile). 4 waves; wave w owns channels
//    [128w,128w+128) of V/O. Online softmax state (m,l) in LDS.
// ---------------------------------------------------------------------------
__global__ void __launch_bounds__(128)
attn_kernel(const bf16_t* __restrict__ qm, const bf16_t* __restrict__ km,
            const bf16_t* __restrict__ vT, const bf16_t* __restrict__ wpT,
            const float* __restrict__ bp, const float* __restrict__ x,
            float* __restrict__ out) {
  __shared__ __align__(16) bf16_t q_lds[16][C_];   // 16 KB
  __shared__ __align__(16) bf16_t o_lds[16][C_];   // 16 KB
  __shared__ __align__(16) bf16_t p_lds[16][64];   // 2 KB
  __shared__ float smax[4][16];
  __shared__ float ssum[4][16];
  __shared__ float m_state[16], l_state[16], corr_s[16];

  int tid  = threadIdx.x;
  int lane = tid & 31;
  int wave = tid >> 5;
  int b  = blockIdx.x >> 6;
  int qb = blockIdx.x & 63;
  int m0 = qb * 16;
  int lm    = lane & 15;
  int half  = lane >> 4;
  int half8 = half * 8;
  int koffA = half * 16;
  int koffB = half * 16;

  // async-stage Q tile (16x512 bf16 = 16 KB): 8 x b128 per thread
  const bf16_t* qbase = qm + ((size_t)b * N_ + m0) * C_;
#pragma unroll
  for (int it = 0; it < 8; ++it) {
    int i = it * 128 + tid;
    int row = i >> 6;                 // (i*8)/512
    int col = (i * 8) & 511;
    async_load_b128(&q_lds[row][col], qbase + (size_t)row * C_ + col);
  }
  if (tid < 16) { m_state[tid] = -3.0e38f; l_state[tid] = 0.f; }
  wait_async0();
  __syncthreads();

  v8f oacc[8] = {};

  for (int blk = 0; blk < 16; ++blk) {          // 64 keys per block
    int kb = blk * 64;
    int kw = kb + wave * 16;                    // this wave's 16 keys

    // ---- S = Q @ K^T (scale already folded into Q) : 16 WMMA
    v8f s = {};
    const bf16_t* krow = km + ((size_t)b * N_ + kw + lm) * C_ + koffB;
#pragma unroll
    for (int kk = 0; kk < C_; kk += 32) {
      V16 a;
      a.h[0] = *(const v8bf*)&q_lds[lm][kk + koffA];
      a.h[1] = *(const v8bf*)&q_lds[lm][kk + koffA + 8];
      v16bf bm = *(const v16bf*)(krow + kk);
      s = wmma_bf16(a.v, bm, s);
    }

    // ---- prefetch next block's K tile while VMEM is idle (softmax phase)
    if (blk + 1 < 16) {
      const bf16_t* nk = km + ((size_t)b * N_ + kb + 64 + (tid & 63)) * C_ + (tid >> 6) * 256;
      __builtin_prefetch(nk, 0, 1);
      __builtin_prefetch(nk + 128, 0, 1);
    }

    // ---- per-row max of this 16x16 tile (rows live in 16-lane halves)
    float rmax[8];
#pragma unroll
    for (int r = 0; r < 8; ++r) rmax[r] = s[r];
#pragma unroll
    for (int off = 8; off > 0; off >>= 1)
#pragma unroll
      for (int r = 0; r < 8; ++r)
        rmax[r] = fmaxf(rmax[r], __shfl_xor(rmax[r], off, 32));
    if (lm == 0) {
#pragma unroll
      for (int r = 0; r < 8; ++r) smax[wave][r + half8] = rmax[r];
    }
    __syncthreads();                               // smax ready

    // ---- new running max per owned row (redundant per lane)
    float mn[8];
#pragma unroll
    for (int r = 0; r < 8; ++r) {
      int row = r + half8;
      float m = m_state[row];
      m = fmaxf(m, smax[0][row]); m = fmaxf(m, smax[1][row]);
      m = fmaxf(m, smax[2][row]); m = fmaxf(m, smax[3][row]);
      mn[r] = m;
    }

    // ---- P = exp(S - m_new); row sums; stage P (bf16) to LDS
    float rsum[8];
#pragma unroll
    for (int r = 0; r < 8; ++r) {
      float p = __expf(s[r] - mn[r]);
      rsum[r] = p;
      p_lds[r + half8][wave * 16 + lm] = (bf16_t)p;
    }
#pragma unroll
    for (int off = 8; off > 0; off >>= 1)
#pragma unroll
      for (int r = 0; r < 8; ++r)
        rsum[r] += __shfl_xor(rsum[r], off, 32);
    if (lm == 0) {
#pragma unroll
      for (int r = 0; r < 8; ++r) ssum[wave][r + half8] = rsum[r];
    }
    __syncthreads();                               // P + ssum ready

    // ---- update online-softmax state (wave 0, one lane per row)
    if (wave == 0 && lane < 16) {
      int row = lane;
      float mo = m_state[row];
      float m2 = fmaxf(fmaxf(smax[0][row], smax[1][row]),
                       fmaxf(smax[2][row], smax[3][row]));
      m2 = fmaxf(m2, mo);
      float cfac = __expf(mo - m2);
      l_state[row] = l_state[row] * cfac +
                     ssum[0][row] + ssum[1][row] + ssum[2][row] + ssum[3][row];
      m_state[row] = m2;
      corr_s[row]  = cfac;
    }
    __syncthreads();                               // corr ready

    // ---- rescale accumulators, then O += P @ V (16 WMMA)
    float cf[8];
#pragma unroll
    for (int r = 0; r < 8; ++r) cf[r] = corr_s[r + half8];
#pragma unroll
    for (int t = 0; t < 8; ++t)
#pragma unroll
      for (int r = 0; r < 8; ++r) oacc[t][r] *= cf[r];

#pragma unroll
    for (int kt = 0; kt < 2; ++kt) {
      V16 a;
      a.h[0] = *(const v8bf*)&p_lds[lm][kt * 32 + koffA];
      a.h[1] = *(const v8bf*)&p_lds[lm][kt * 32 + koffA + 8];
#pragma unroll
      for (int t = 0; t < 8; ++t) {
        int col = wave * 128 + t * 16 + lm;
        v16bf bm = *(const v16bf*)(vT + ((size_t)b * C_ + col) * N_ + kb + kt * 32 + koffB);
        oacc[t] = wmma_bf16(a.v, bm, oacc[t]);
      }
    }
  }

  // ---- finalize softmax (divide by l), stage O (bf16) to LDS
  float li[8];
#pragma unroll
  for (int r = 0; r < 8; ++r) li[r] = 1.0f / l_state[r + half8];
#pragma unroll
  for (int t = 0; t < 8; ++t) {
    int col = wave * 128 + t * 16 + lm;
#pragma unroll
    for (int r = 0; r < 8; ++r)
      o_lds[r + half8][col] = (bf16_t)(oacc[t][r] * li[r]);
  }
  __syncthreads();

  // ---- output projection: OUT = O @ wp + bp + residual (128 WMMA / wave)
  v8f pacc[8] = {};
#pragma unroll
  for (int kk = 0; kk < C_; kk += 32) {
    V16 a;
    a.h[0] = *(const v8bf*)&o_lds[lm][kk + koffA];
    a.h[1] = *(const v8bf*)&o_lds[lm][kk + koffA + 8];
#pragma unroll
    for (int t = 0; t < 8; ++t) {
      int col = wave * 128 + t * 16 + lm;
      v16bf bm = *(const v16bf*)(wpT + (size_t)col * C_ + kk + koffB);
      pacc[t] = wmma_bf16(a.v, bm, pacc[t]);
    }
  }
#pragma unroll
  for (int t = 0; t < 8; ++t) {
    int col = wave * 128 + t * 16 + lm;
    float bpv = bp[col];
#pragma unroll
    for (int r = 0; r < 8; ++r) {
      size_t idx = ((size_t)b * N_ + m0 + r + half8) * C_ + col;
      out[idx] = pacc[t][r] + bpv + x[idx];
    }
  }
}

// ---------------------------------------------------------------------------
extern "C" void kernel_launch(void* const* d_in, const int* in_sizes, int n_in,
                              void* d_out, int out_size, void* d_ws, size_t ws_size,
                              hipStream_t stream) {
  const float* x     = (const float*)d_in[0];
  const float* gamma = (const float*)d_in[1];
  const float* beta  = (const float*)d_in[2];
  const float* wq    = (const float*)d_in[3];
  const float* bq    = (const float*)d_in[4];
  const float* wk    = (const float*)d_in[5];
  const float* bk    = (const float*)d_in[6];
  const float* wv    = (const float*)d_in[7];
  const float* bv    = (const float*)d_in[8];
  const float* wp    = (const float*)d_in[9];
  const float* bp    = (const float*)d_in[10];
  float* out = (float*)d_out;

  char* ws = (char*)d_ws;
  const size_t oW = (size_t)C_ * C_ * sizeof(bf16_t);        // 512 KB / weight
  const size_t oX = (size_t)B_ * N_ * C_ * sizeof(bf16_t);   // 16 MB / tensor
  bf16_t* wqT = (bf16_t*)(ws + 0 * oW);
  bf16_t* wkT = (bf16_t*)(ws + 1 * oW);
  bf16_t* wvT = (bf16_t*)(ws + 2 * oW);
  bf16_t* wpT = (bf16_t*)(ws + 3 * oW);
  bf16_t* xn  = (bf16_t*)(ws + 4 * oW);
  bf16_t* qm  = (bf16_t*)(ws + 4 * oW + 1 * oX);
  bf16_t* km  = (bf16_t*)(ws + 4 * oW + 2 * oX);
  bf16_t* vT  = (bf16_t*)(ws + 4 * oW + 3 * oX);             // total ~66 MB

  const float qscale = 0.044194173824159216f;                // 512^-0.5

  cvt_weights_kernel<<<dim3(1024, 4), 256, 0, stream>>>(wq, wk, wv, wp, wqT, wkT, wvT, wpT);
  groupnorm_kernel<<<B_ * G_, 256, 0, stream>>>(x, gamma, beta, xn);
  qkv_gemm_kernel<<<1024, 256, 0, stream>>>(xn, wqT, bq, qm, qscale, 0);
  qkv_gemm_kernel<<<1024, 256, 0, stream>>>(xn, wkT, bk, km, 1.0f, 0);
  qkv_gemm_kernel<<<1024, 256, 0, stream>>>(xn, wvT, bv, vT, 1.0f, 1);
  attn_kernel<<<B_ * (N_ / 16), 128, 0, stream>>>(qm, km, vT, wpT, bp, x, out);
}
